// BasicUpdateBlock_7713761263651
// MI455X (gfx1250) — matrix operand order repaired and checked by reading, not compiled, not run on hardware
//
#include <hip/hip_runtime.h>
#include <hip/hip_bf16.h>

typedef _Float16 half_t;
typedef __attribute__((ext_vector_type(16))) _Float16 v16h;
typedef __attribute__((ext_vector_type(8)))  _Float16 v8h;
typedef __attribute__((ext_vector_type(8)))  float    v8f;

// Fixed problem geometry (reference: B=4, H=W=512) -> all addressing is
// shifts/masks, no runtime division in image (H/W) addressing.
constexpr int IMG_B = 4;
constexpr int IMG_H = 512;
constexpr int IMG_W = 512;
constexpr int IMG_HW = IMG_H * IMG_W;        // 2^18

__device__ __forceinline__ int imin(int a, int b) { return a < b ? a : b; }
__device__ __forceinline__ int imax(int a, int b) { return a > b ? a : b; }

// ---------------------------------------------------------------------------
// Weight conversion: f32 OIHW -> f16 [CoPad][Kpad], zero padded.
// k = ci*KH*KW + ky*KW + kx  (exactly the OIHW flattening)
// ---------------------------------------------------------------------------
__global__ void wcvt_kernel(const float* __restrict__ w, half_t* __restrict__ o,
                            int Cout, int K, int CoPad, int Kpad) {
  int i = blockIdx.x * blockDim.x + threadIdx.x;
  int tot = CoPad * Kpad;
  if (i >= tot) return;
  int co = i / Kpad, k = i - co * Kpad;
  float v = (co < Cout && k < K) ? w[(size_t)co * K + k] : 0.0f;
  o[i] = (half_t)v;
}

// ---------------------------------------------------------------------------
// Generic implicit-GEMM conv via v_wmma_f32_16x16x32_f16, templated on the
// window so all im2col index math uses compile-time divisors + an
// increment-with-carry decode (no runtime division in the hot loop).
// Each wave: 16 output channels x 16 consecutive pixels (one row; W%16==0).
// Input may be a virtual concat of two tensors: channels [0,CA) from inA,
// [CA, CA+CB) from inB. Activation fused in epilogue.
// act: 0=none 1=relu 2=leaky(0.2) 3=sigmoid 4=tanh
// ---------------------------------------------------------------------------
template <int KH, int KW>
__global__ __launch_bounds__(256)
void conv_wmma_kernel(const float* __restrict__ inA, int CA,
                      const float* __restrict__ inB, int CB,
                      const half_t* __restrict__ Wh, const float* __restrict__ bias,
                      float* __restrict__ out, int outCt, int outCo, int Cout,
                      int pad, int Ktot, int Kpad, int act) {
  constexpr int KHW = KH * KW;
  const int lane = threadIdx.x & 31;
  const int wave = threadIdx.x >> 5;
  const int pTile = blockIdx.x * 8 + wave;
  const int coTile = blockIdx.y;
  const int n  = lane & 15;
  const bool hi = (lane & 16) != 0;
  const int p  = pTile * 16 + n;
  const int b  = p >> 18;                 // / IMG_HW
  const int rem = p & (IMG_HW - 1);
  const int y  = rem >> 9;                // / IMG_W
  const int x  = rem & (IMG_W - 1);
  const half_t* wrow = Wh + (size_t)(coTile * 16 + n) * Kpad;

  v8f c = {0.f, 0.f, 0.f, 0.f, 0.f, 0.f, 0.f, 0.f};

  for (int k0 = 0; k0 < Kpad; k0 += 32) {
    // ---- A operand: 16x32, contiguous k order (same as B) ----
    // lanes 0-15: halves 0..15 = K k0..k0+15
    // lanes16-31: halves 0..15 = K k0+16..k0+31
    int ka = k0 + (hi ? 16 : 0);
    v8h a0 = *(const v8h*)(wrow + ka);
    v8h a1 = *(const v8h*)(wrow + ka + 8);
    v16h a;
#pragma unroll
    for (int i = 0; i < 8; ++i) { a[i] = a0[i]; a[8 + i] = a1[i]; }

    // ---- B operand: 32x16, lanes 0-15 hold K k0..k0+15, lanes 16-31 +16 ----
    const int kb = k0 + (hi ? 16 : 0);
    // one constant-divisor decode per chunk, then increment-with-carry
    int ci = kb / KHW;
    int r  = kb - ci * KHW;
    int ky = r / KW;
    int kx = r - ky * KW;
    const bool full = (kb + 16 <= Ktot);   // whole chunk-half in range
    v16h bm;
#pragma unroll
    for (int j = 0; j < 16; ++j) {
      float v = 0.0f;
      if (full || (kb + j < Ktot)) {
        int yy = y + ky - pad;
        int xx = x + kx - pad;
        if (yy >= 0 && yy < IMG_H && xx >= 0 && xx < IMG_W) {
          const float* src = (ci < CA) ? inA : inB;
          int cc = (ci < CA) ? ci : (ci - CA);
          int Ct = (ci < CA) ? CA : CB;
          v = src[(((size_t)b * Ct + cc) << 18) + (yy << 9) + xx];
        }
      }
      bm[j] = (half_t)v;
      if (++kx == KW) { kx = 0; if (++ky == KH) { ky = 0; ++ci; } }
    }
    c = __builtin_amdgcn_wmma_f32_16x16x32_f16(false, a, false, bm,
                                               (short)0, c, false, false);
  }

  // ---- Epilogue: C/D layout: c[i] lanes0-15 -> M=i; lanes16-31 -> M=i+8 ----
#pragma unroll
  for (int i = 0; i < 8; ++i) {
    int mr = hi ? (8 + i) : i;
    int co = coTile * 16 + mr;
    if (co < Cout) {
      float v = c[i] + (bias ? bias[co] : 0.0f);
      if      (act == 1) v = v > 0.f ? v : 0.f;
      else if (act == 2) v = v > 0.f ? v : 0.2f * v;
      else if (act == 3) v = 1.0f / (1.0f + __expf(-v));
      else if (act == 4) v = tanhf(v);
      out[(((size_t)b * outCt + outCo + co) << 18) + (y << 9) + x] = v;
    }
  }
}

// ---------------------------------------------------------------------------
// Channel-slice copy into a wider tensor.
// ---------------------------------------------------------------------------
__global__ void copy_chan_kernel(const float* __restrict__ src, int srcCt, int srcCo,
                                 float* __restrict__ dst, int dstCt, int dstCo, int C) {
  size_t idx = (size_t)blockIdx.x * blockDim.x + threadIdx.x;
  size_t total = (size_t)IMG_B * C * IMG_HW;
  if (idx >= total) return;
  int pix = (int)(idx & (IMG_HW - 1));
  int t = (int)(idx >> 18);
  int cch = t % C, b = t / C;
  dst[(((size_t)b * dstCt + dstCo + cch) << 18) + pix] =
      src[(((size_t)b * srcCt + srcCo + cch) << 18) + pix];
}

// ---------------------------------------------------------------------------
// grid_sample warp of h(=net, 16ch) by 4 flows (bilinear, zeros, no align_corners)
// wrapped channel = l*16 + c
// ---------------------------------------------------------------------------
__global__ void warp_kernel(const float* __restrict__ img,
                            const float* __restrict__ flows,
                            float* __restrict__ wrapped) {
  int idx = blockIdx.x * blockDim.x + threadIdx.x;
  int total = IMG_B * 4 * IMG_HW;
  if (idx >= total) return;
  int pix = idx & (IMG_HW - 1);
  int t = idx >> 18;
  int l = t & 3, b = t >> 2;
  int y = pix >> 9, x = pix & (IMG_W - 1);
  float fx = flows[(((size_t)b * 8 + 2 * l) << 18) + pix];
  float fy = flows[(((size_t)b * 8 + 2 * l + 1) << 18) + pix];
  float vgx = (float)x - fx;
  float vgy = (float)y - fy;
  float gx = 2.0f * vgx / (float)(IMG_W - 1) - 1.0f;
  float gy = 2.0f * vgy / (float)(IMG_H - 1) - 1.0f;
  float ix = ((gx + 1.0f) * (float)IMG_W - 1.0f) * 0.5f;
  float iy = ((gy + 1.0f) * (float)IMG_H - 1.0f) * 0.5f;
  float x0f = floorf(ix), y0f = floorf(iy);
  float wx1 = ix - x0f, wy1 = iy - y0f;
  float wx0 = 1.0f - wx1, wy0 = 1.0f - wy1;
  int xi0 = (int)x0f, yi0 = (int)y0f;
  int cx[4] = {xi0, xi0 + 1, xi0, xi0 + 1};
  int cy[4] = {yi0, yi0, yi0 + 1, yi0 + 1};
  float cw[4] = {wx0 * wy0, wx1 * wy0, wx0 * wy1, wx1 * wy1};
  bool cv[4]; int ccx[4], ccy[4];
#pragma unroll
  for (int i = 0; i < 4; ++i) {
    cv[i] = (cx[i] >= 0) && (cx[i] <= IMG_W - 1) && (cy[i] >= 0) && (cy[i] <= IMG_H - 1);
    ccx[i] = imin(imax(cx[i], 0), IMG_W - 1);
    ccy[i] = imin(imax(cy[i], 0), IMG_H - 1);
  }
  for (int c = 0; c < 16; ++c) {
    float acc = 0.f;
#pragma unroll
    for (int i = 0; i < 4; ++i) {
      float v = cv[i] ? img[(((size_t)b * 16 + c) << 18) + (ccy[i] << 9) + ccx[i]] : 0.f;
      acc += v * cw[i];
    }
    wrapped[(((size_t)b * 64 + l * 16 + c) << 18) + pix] = acc;
  }
}

// rh = rg * net
__global__ void mul_kernel(const float* __restrict__ a, const float* __restrict__ b,
                           float* __restrict__ o, size_t n) {
  size_t i = (size_t)blockIdx.x * blockDim.x + threadIdx.x;
  if (i < n) o[i] = a[i] * b[i];
}

// h = (1-zg)*net + zg*q
__global__ void hupdate_kernel(const float* __restrict__ zg, const float* __restrict__ net,
                               const float* __restrict__ q, float* __restrict__ h, size_t n) {
  size_t i = (size_t)blockIdx.x * blockDim.x + threadIdx.x;
  if (i < n) h[i] = (1.0f - zg[i]) * net[i] + zg[i] * q[i];
}

// per-(b,c) spatial mean & max; one block per plane
__global__ void chanstat_kernel(const float* __restrict__ h, float* __restrict__ avg,
                                float* __restrict__ mx) {
  __shared__ float ssum[256], smax[256];
  const float* p = h + ((size_t)blockIdx.x << 18);
  float s = 0.f, m = -3.4e38f;
  for (int i = threadIdx.x; i < IMG_HW; i += 256) { float v = p[i]; s += v; m = fmaxf(m, v); }
  ssum[threadIdx.x] = s; smax[threadIdx.x] = m;
  __syncthreads();
  for (int st = 128; st > 0; st >>= 1) {
    if ((int)threadIdx.x < st) {
      ssum[threadIdx.x] += ssum[threadIdx.x + st];
      smax[threadIdx.x] = fmaxf(smax[threadIdx.x], smax[threadIdx.x + st]);
    }
    __syncthreads();
  }
  if (threadIdx.x == 0) {
    avg[blockIdx.x] = ssum[0] * (1.0f / (float)IMG_HW);
    mx[blockIdx.x] = smax[0];
  }
}

// att[b,c] = sigmoid(w2[c]*relu(dot(w1,avg[b,:])) + w2[c]*relu(dot(w1,mx[b,:])))
__global__ void chanatt_kernel(const float* __restrict__ avg, const float* __restrict__ mx,
                               const float* __restrict__ w1, const float* __restrict__ w2,
                               float* __restrict__ att) {
  __shared__ float mavg[8], mmax[8];
  int t = threadIdx.x;
  if (t < IMG_B) {
    float sa = 0.f, sm = 0.f;
    for (int c = 0; c < 16; ++c) { sa += w1[c] * avg[t * 16 + c]; sm += w1[c] * mx[t * 16 + c]; }
    mavg[t] = fmaxf(sa, 0.f); mmax[t] = fmaxf(sm, 0.f);
  }
  __syncthreads();
  if (t < IMG_B * 16) {
    int b = t >> 4, c = t & 15;
    float v = w2[c] * mavg[b] + w2[c] * mmax[b];
    att[t] = 1.0f / (1.0f + __expf(-v));
  }
}

__global__ void chanscale_kernel(float* __restrict__ h, const float* __restrict__ att,
                                 size_t n) {
  size_t i = (size_t)blockIdx.x * blockDim.x + threadIdx.x;
  if (i < n) h[i] *= att[i >> 18];
}

// sa2[b,0]=mean_c h, sa2[b,1]=max_c h
__global__ void spatstat_kernel(const float* __restrict__ h, float* __restrict__ sa2) {
  size_t idx = (size_t)blockIdx.x * blockDim.x + threadIdx.x;
  size_t total = (size_t)IMG_B * IMG_HW;
  if (idx >= total) return;
  int b = (int)(idx >> 18);
  int pix = (int)(idx & (IMG_HW - 1));
  float s = 0.f, m = -3.4e38f;
  for (int c = 0; c < 16; ++c) {
    float v = h[(((size_t)b * 16 + c) << 18) + pix];
    s += v; m = fmaxf(m, v);
  }
  sa2[(((size_t)b * 2 + 0) << 18) + pix] = s * (1.0f / 16.0f);
  sa2[(((size_t)b * 2 + 1) << 18) + pix] = m;
}

// h *= sa(b,pix)  (sa conv already applied sigmoid)
__global__ void spatscale_kernel(float* __restrict__ h, const float* __restrict__ sa,
                                 size_t n) {
  size_t i = (size_t)blockIdx.x * blockDim.x + threadIdx.x;
  if (i >= n) return;
  int pix = (int)(i & (IMG_HW - 1));
  int b = (int)((i >> 18) >> 4);
  h[i] *= sa[((size_t)b << 18) + pix];
}

// ---------------------------------------------------------------------------
extern "C" void kernel_launch(void* const* d_in, const int* in_sizes, int n_in,
                              void* d_out, int out_size, void* d_ws, size_t ws_size,
                              hipStream_t stream) {
  const int P = IMG_B * IMG_HW;       // 1,048,576 pixels (divisible by 128)
  auto SZ = [&](int C) { return (size_t)IMG_B * C * IMG_HW; };

  const float* net   = (const float*)d_in[0];
  const float* inp   = (const float*)d_in[1];
  const float* corr  = (const float*)d_in[2];
  const float* noise = (const float*)d_in[3];
  const float* wc1 = (const float*)d_in[4];  const float* bc1 = (const float*)d_in[5];
  const float* wc2 = (const float*)d_in[6];  const float* bc2 = (const float*)d_in[7];
  const float* wf1 = (const float*)d_in[8];  const float* bf1 = (const float*)d_in[9];
  const float* wf2 = (const float*)d_in[10]; const float* bf2 = (const float*)d_in[11];
  const float* we  = (const float*)d_in[12]; const float* be  = (const float*)d_in[13];
  const float* loc1w = (const float*)d_in[14]; const float* loc1b = (const float*)d_in[15];
  const float* loc2w = (const float*)d_in[16]; const float* loc2b = (const float*)d_in[17];
  const float* wz = (const float*)d_in[18]; const float* bz = (const float*)d_in[19];
  const float* wr = (const float*)d_in[20]; const float* br = (const float*)d_in[21];
  const float* wq = (const float*)d_in[22]; const float* bq = (const float*)d_in[23];
  const float* caw1 = (const float*)d_in[24];
  const float* caw2 = (const float*)d_in[25];
  const float* saw  = (const float*)d_in[26];
  const float* fh1w = (const float*)d_in[27]; const float* fh1b = (const float*)d_in[28];
  const float* fh2w = (const float*)d_in[29]; const float* fh2b = (const float*)d_in[30];

  float* ws = (float*)d_ws;
  // workspace layout (floats), with region reuse
  float* tmp32 = ws;                         // SZ(32): conv scratch; later flows/q/sa
  float* catb  = ws + SZ(32);                // SZ(64): [cor|flo]; later t/zg/rg/rh; later fh1out
  float* Xb    = catb + SZ(64);              // SZ(80): x = [inp|mo|noise]
  float* wrapb = Xb + SZ(80);                // SZ(64): wrapped
  float* small = wrapb + SZ(64);             // stats: avg(64) mx(64) att(64)
  half_t* whb  = (half_t*)(small + 256);     // f16 weight region (~360 KB)

  float* flows = tmp32;                      // SZ(8)   (after wf2 done)
  float* qbuf  = tmp32 + SZ(8);              // SZ(16)
  float* sa2   = tmp32;                      // SZ(2)   (after flows/q dead)
  float* saout = tmp32 + SZ(2);              // SZ(1)
  float* tbuf  = catb;                       // SZ(16)  (after cat64 dead)
  float* zg    = catb + SZ(16);              // SZ(16)
  float* rg    = catb + SZ(32);              // SZ(16)
  float* rh    = catb + SZ(48);              // SZ(16)
  float* fh1o  = catb;                       // SZ(64)  (after zg/rg/rh dead)
  float* avgp = small, *mxp = small + 64, *attp = small + 128;

  float* hout  = (float*)d_out;              // (B,16,H,W)
  float* delta = hout + SZ(16);              // (B,3,H,W)

  // ---- f16 weight conversion (padded [CoPad][Kpad]) ----
  struct WInfo { const float* w; int Cout, K; size_t off; };
  auto copad = [](int c) { return ((c + 15) / 16) * 16; };
  auto kpad  = [](int k) { return (k + 31) & ~31; };
  size_t off = 0;
  WInfo wi[13];
  auto addw = [&](int i, const float* w, int Cout, int K) {
    wi[i] = {w, Cout, K, off};
    off += (size_t)copad(Cout) * kpad(K);
  };
  addw(0,  wc1,  32, 27);    addw(1,  wc2,  32, 288);
  addw(2,  wf1,  32, 27);    addw(3,  wf2,  32, 288);
  addw(4,  we,   61, 576);   addw(5,  loc1w, 16, 2400);
  addw(6,  loc2w, 8, 400);   addw(7,  wz,   16, 1296);
  addw(8,  wr,   16, 1296);  addw(9,  wq,   16, 864);
  addw(10, saw,   1, 98);    addw(11, fh1w, 64, 144);
  addw(12, fh2w,  3, 576);
  for (int i = 0; i < 13; ++i) {
    int tot = copad(wi[i].Cout) * kpad(wi[i].K);
    wcvt_kernel<<<(tot + 255) / 256, 256, 0, stream>>>(
        wi[i].w, whb + wi[i].off, wi[i].Cout, wi[i].K, copad(wi[i].Cout), kpad(wi[i].K));
  }

  auto conv = [&](const float* inA, int CA, const float* inB, int CBc, int widx,
                  const float* bias, float* out, int outCt, int outCo, int Cout,
                  int KH, int pad, int act) {
    int Ktot = (CA + CBc) * KH * KH;
    dim3 grid(P / 128, (Cout + 15) / 16);
    if (KH == 3)
      conv_wmma_kernel<3, 3><<<grid, 256, 0, stream>>>(
          inA, CA, inB, CBc, whb + wi[widx].off, bias, out, outCt, outCo, Cout,
          pad, Ktot, kpad(Ktot), act);
    else if (KH == 5)
      conv_wmma_kernel<5, 5><<<grid, 256, 0, stream>>>(
          inA, CA, inB, CBc, whb + wi[widx].off, bias, out, outCt, outCo, Cout,
          pad, Ktot, kpad(Ktot), act);
    else
      conv_wmma_kernel<7, 7><<<grid, 256, 0, stream>>>(
          inA, CA, inB, CBc, whb + wi[widx].off, bias, out, outCt, outCo, Cout,
          pad, Ktot, kpad(Ktot), act);
  };
  auto eltgrid = [](size_t n) { return dim3((unsigned)((n + 255) / 256)); };

  // ---- BasicMotionEncoder ----
  conv(corr, 3, nullptr, 0, 0, bc1, tmp32, 32, 0, 32, 3, 1, 1);          // cor1
  conv(tmp32, 32, nullptr, 0, 1, bc2, catb, 64, 0, 32, 3, 1, 1);         // cor -> cat[0:32]
  conv(noise, 3, nullptr, 0, 2, bf1, tmp32, 32, 0, 32, 3, 1, 1);         // flo1
  conv(tmp32, 32, nullptr, 0, 3, bf2, catb, 64, 32, 32, 3, 1, 1);        // flo -> cat[32:64]
  conv(catb, 64, nullptr, 0, 4, be, Xb, 80, 16, 61, 3, 1, 1);            // mo -> X[16:77]
  copy_chan_kernel<<<eltgrid(SZ(16)), 256, 0, stream>>>(inp, 16, 0, Xb, 80, 0, 16);
  copy_chan_kernel<<<eltgrid(SZ(3)), 256, 0, stream>>>(noise, 3, 0, Xb, 80, 77, 3);

  // ---- local flows ----
  conv(Xb, 80, net, 16, 5, loc1b, tbuf, 16, 0, 16, 5, 2, 2);             // leaky 0.2
  conv(tbuf, 16, nullptr, 0, 6, loc2b, flows, 8, 0, 8, 5, 2, 0);         // flows

  // ---- warp h by 4 flows ----
  warp_kernel<<<eltgrid((size_t)IMG_B * 4 * IMG_HW), 256, 0, stream>>>(net, flows, wrapb);

  // ---- GRU gates ----
  conv(wrapb, 64, Xb, 80, 7, bz, zg, 16, 0, 16, 3, 1, 3);                // sigmoid
  conv(wrapb, 64, Xb, 80, 8, br, rg, 16, 0, 16, 3, 1, 3);                // sigmoid
  mul_kernel<<<eltgrid(SZ(16)), 256, 0, stream>>>(rg, net, rh, SZ(16));
  conv(rh, 16, Xb, 80, 9, bq, qbuf, 16, 0, 16, 3, 1, 4);                 // tanh
  hupdate_kernel<<<eltgrid(SZ(16)), 256, 0, stream>>>(zg, net, qbuf, hout, SZ(16));

  // ---- CBAM channel attention ----
  chanstat_kernel<<<IMG_B * 16, 256, 0, stream>>>(hout, avgp, mxp);
  chanatt_kernel<<<1, 64, 0, stream>>>(avgp, mxp, caw1, caw2, attp);
  chanscale_kernel<<<eltgrid(SZ(16)), 256, 0, stream>>>(hout, attp, SZ(16));

  // ---- CBAM spatial attention ----
  spatstat_kernel<<<eltgrid((size_t)IMG_B * IMG_HW), 256, 0, stream>>>(hout, sa2);
  conv(sa2, 2, nullptr, 0, 10, nullptr, saout, 1, 0, 1, 7, 3, 3);        // sigmoid
  spatscale_kernel<<<eltgrid(SZ(16)), 256, 0, stream>>>(hout, saout, SZ(16));

  // ---- FlowHead ----
  conv(hout, 16, nullptr, 0, 11, fh1b, fh1o, 64, 0, 64, 3, 1, 1);        // relu
  conv(fh1o, 64, nullptr, 0, 12, fh2b, delta, 3, 0, 3, 3, 1, 0);         // delta_flow
}
